// NodeLevel_DFM_DiffusionNet_37641093382400
// MI455X (gfx1250) — matrix operand order repaired and checked by reading, not compiled, not run on hardware
//
#include <hip/hip_runtime.h>
#include <hip/hip_bf16.h>

#define NB    8
#define NPG   12500
#define NN    100000
#define KK    128
#define CC    128
#define IND   7
#define GSD   11
#define NBLKS 4
#define NCLF  5
#define EE    800000
#define RBPG  98          // ceil(12500/128) row-blocks per graph
#define RBALL 782         // ceil(100000/128) row-blocks over all nodes

typedef __attribute__((ext_vector_type(16))) __bf16 bf16x16;
typedef __attribute__((ext_vector_type(8)))  __bf16 bf16x8;
typedef __attribute__((ext_vector_type(2)))  __bf16 bf16x2;
typedef __attribute__((ext_vector_type(8)))  float  f32x8;
typedef __attribute__((ext_vector_type(2)))  float  f32x2;

union Frag16 { uint4 u[2]; bf16x16 v; };

__device__ __forceinline__ unsigned short bfbits(float f) {
  union { __bf16 h; unsigned short s; } r;
  r.h = (__bf16)f;                       // native cvt, RNE
  return r.s;
}

__device__ __forceinline__ unsigned int pk2(float a, float b) {
  f32x2 v = {a, b};
  union { bf16x2 h; unsigned int u; } r;
  r.h = __builtin_convertvector(v, bf16x2);
  return r.u;
}

// 8 f32 * m -> 8 bf16 packed in a uint4 (native packed cvt path)
__device__ __forceinline__ uint4 cvt16(float4 a, float4 b, float m) {
  f32x8 v = {a.x * m, a.y * m, a.z * m, a.w * m, b.x * m, b.y * m, b.z * m, b.w * m};
  union { bf16x8 h; uint4 u; } r;
  r.h = __builtin_convertvector(v, bf16x8);
  return r.u;
}

// Convert 16 contiguous f32 (16B-aligned src) * m -> 16 bf16 -> two ds_store_b128.
__device__ __forceinline__ void stage16(unsigned short* dst, const float* src, float m) {
  const float4* s4 = (const float4*)src;
  float4 f0 = s4[0], f1 = s4[1], f2 = s4[2], f3 = s4[3];
  *(uint4*)(dst)     = cvt16(f0, f1, m);
  *(uint4*)(dst + 8) = cvt16(f2, f3, m);
}

// A fragment (16x32 bf16, M x K). LDS tile row-major [16][stride]; lane m = l&15;
// K halves {0..7,16..23} (lanes 0-15) / {8..15,24..31} (lanes 16-31).
__device__ __forceinline__ bf16x16 frag_a_lds(const unsigned short* lds, int stride) {
  int l = threadIdx.x & 31;
  const unsigned short* p = lds + (l & 15) * stride + ((l < 16) ? 0 : 8);
  Frag16 r;
  r.u[0] = *(const uint4*)(p);        // K = base..base+7
  r.u[1] = *(const uint4*)(p + 16);   // K = base+16..base+23
  return r.v;
}

// B fragment (32x16 bf16, K x N). LDS tile row-major [32][stride]; lane n = l&15;
// K halves {0..7,16..23} (lanes 0-15) / {8..15,24..31} (lanes 16-31).
__device__ __forceinline__ bf16x16 frag_b_lds(const unsigned short* lds, int stride) {
  int l = threadIdx.x & 31;
  const unsigned short* p = lds + ((l < 16) ? 0 : 8) * stride + (l & 15);
  union { unsigned short s[16]; bf16x16 v; } r;
#pragma unroll
  for (int s = 0; s < 8; ++s) {
    r.s[s] = p[s * stride];              // K = base..base+7
    r.s[s + 8] = p[(s + 16) * stride];   // K = base+16..base+23
  }
  return r.v;
}

#define WMMA_BF16(a, b, c) \
  __builtin_amdgcn_wmma_f32_16x16x32_bf16(false, (a), false, (b), (short)0, (c), false, false)

__device__ __forceinline__ void atomAddF(float* p, float v) {
  __hip_atomic_fetch_add(p, v, __ATOMIC_RELAXED, __HIP_MEMORY_SCOPE_AGENT);
}

__device__ __forceinline__ int rfl(int v) { return __builtin_amdgcn_readfirstlane(v); }

// ---------------- embed: h = relu(x @ W_embed + b_embed), K=7 (scalar, mem-bound) --------
__global__ void __launch_bounds__(256) k_embed(const float* __restrict__ x,
                                               const float* __restrict__ W,
                                               const float* __restrict__ b,
                                               float* __restrict__ h) {
  int idx = blockIdx.x * blockDim.x + threadIdx.x;
  if (idx >= NN * CC) return;
  int n = idx >> 7, c = idx & 127;
  float s = b[c];
  const float* xr = x + n * IND;
#pragma unroll
  for (int i = 0; i < IND; ++i) s += xr[i] * W[i * CC + c];
  h[idx] = fmaxf(s, 0.f);
}

__global__ void __launch_bounds__(256) k_fill0(float* __restrict__ p, int count) {
  int idx = blockIdx.x * blockDim.x + threadIdx.x;
  if (idx < count) p[idx] = 0.f;
}

// ---------------- to_basis body (shared full/tail) ---------------------------------------
template <bool MASKED>
__device__ __forceinline__ void tb_body(int n0, int nend, size_t base, int k0, int c0,
                                        const float* __restrict__ h,
                                        const float* __restrict__ evecs,
                                        const float* __restrict__ mass,
                                        unsigned short* la, unsigned short* lb,
                                        f32x8& acc) {
  int lane = threadIdx.x & 31;
  int nn = n0 + lane;
  float msk = 1.f;
  size_t gn;
  if (MASKED) {
    msk = (nn < nend) ? 1.f : 0.f;
    gn = base + ((nn < nend) ? nn : (nend - 1));     // clamped: always valid
  } else {
    gn = base + nn;
  }
  // A^T tile: a[k][lane=n] = evecs[gn*KK + k0 + k] (scattered column write)
  const float4* ep4 = (const float4*)(evecs + gn * KK + k0);
  float4 e0 = ep4[0], e1 = ep4[1], e2 = ep4[2], e3 = ep4[3];
  la[0 * 40 + lane]  = bfbits(e0.x * msk);  la[1 * 40 + lane]  = bfbits(e0.y * msk);
  la[2 * 40 + lane]  = bfbits(e0.z * msk);  la[3 * 40 + lane]  = bfbits(e0.w * msk);
  la[4 * 40 + lane]  = bfbits(e1.x * msk);  la[5 * 40 + lane]  = bfbits(e1.y * msk);
  la[6 * 40 + lane]  = bfbits(e1.z * msk);  la[7 * 40 + lane]  = bfbits(e1.w * msk);
  la[8 * 40 + lane]  = bfbits(e2.x * msk);  la[9 * 40 + lane]  = bfbits(e2.y * msk);
  la[10 * 40 + lane] = bfbits(e2.z * msk);  la[11 * 40 + lane] = bfbits(e2.w * msk);
  la[12 * 40 + lane] = bfbits(e3.x * msk);  la[13 * 40 + lane] = bfbits(e3.y * msk);
  la[14 * 40 + lane] = bfbits(e3.z * msk);  la[15 * 40 + lane] = bfbits(e3.w * msk);
  // B tile: b[lane=n][c] = mass[gn]*h[gn*CC + c0 + c]
  float mm = MASKED ? (mass[gn] * msk) : mass[gn];
  stage16(lb + lane * 24, h + gn * CC + c0, mm);
  bf16x16 a = frag_a_lds(la, 40);
  bf16x16 bb = frag_b_lds(lb, 24);
  acc = WMMA_BF16(a, bb, acc);
}

// grid 512 blocks x 128 thr; wave = n-chunk (4-way split reduction, f32 atomics into spec)
__global__ void __launch_bounds__(128) k_to_basis(const float* __restrict__ h,
                                                  const float* __restrict__ evecs,
                                                  const float* __restrict__ mass,
                                                  float* __restrict__ spec) {
  __shared__ alignas(16) unsigned short sA[4][16 * 40];
  __shared__ alignas(16) unsigned short sB[4][32 * 24];
  int w = rfl(threadIdx.x >> 5), lane = threadIdx.x & 31;
  int tile = blockIdx.x;                      // 0..511 : b*64 + ktile*8 + ctile
  int b = tile >> 6, tt = tile & 63;
  int k0 = (tt >> 3) * 16, c0 = (tt & 7) * 16;
  size_t base = (size_t)b * NPG;
  int nbeg = rfl(w * 3125), nend = rfl(nbeg + 3125);   // 4*3125 == NPG (scalar bounds)
  unsigned short* la = sA[w];
  unsigned short* lb = sB[w];
  f32x8 acc = {0, 0, 0, 0, 0, 0, 0, 0};
  int n0 = nbeg;
  for (; n0 + 32 <= nend; n0 += 32)           // 97 full, branch-free iterations
    tb_body<false>(n0, nend, base, k0, c0, h, evecs, mass, la, lb, acc);
  if (n0 < nend)                               // masked tail (21 lanes)
    tb_body<true>(n0, nend, base, k0, c0, h, evecs, mass, la, lb, acc);
  int nc = lane & 15, hi = (lane < 16) ? 0 : 8;
#pragma unroll
  for (int j = 0; j < 8; ++j) {
    int m = j + hi;
    atomAddF(&spec[((size_t)b * KK + (k0 + m)) * CC + c0 + nc], acc[j]);
  }
}

// ------------- from_basis: h_diff = evecs @ (spec * exp(-evals*time)) --------------------
__device__ __forceinline__ float4 specscale(float4 s, float4 t, float ev) {
  float4 r;
  r.x = s.x * __expf(-ev * fmaxf(fabsf(t.x), 1e-8f));
  r.y = s.y * __expf(-ev * fmaxf(fabsf(t.y), 1e-8f));
  r.z = s.z * __expf(-ev * fmaxf(fabsf(t.z), 1e-8f));
  r.w = s.w * __expf(-ev * fmaxf(fabsf(t.w), 1e-8f));
  return r;
}

__global__ void __launch_bounds__(256) k_from_basis(const float* __restrict__ spec,
                                                    const float* __restrict__ evecs,
                                                    const float* __restrict__ evals,
                                                    const float* __restrict__ tdb,
                                                    float* __restrict__ hd) {
  __shared__ alignas(16) unsigned short sS[KK * 136];
  __shared__ alignas(16) unsigned short sA[8][16 * 40];
  int tid = threadIdx.x;
  int b = blockIdx.x / RBPG, rb = blockIdx.x % RBPG;
  // stage scaled spec as bf16: each thread converts 16 contiguous elems x 4 passes
#pragma unroll
  for (int pass = 0; pass < 4; ++pass) {
    int r = pass * 32 + (tid >> 3);
    int c0 = (tid & 7) * 16;
    float ev = evals[b * KK + r];
    const float4* sp4 = (const float4*)(spec + (size_t)b * KK * CC + (size_t)r * CC + c0);
    const float4* td4 = (const float4*)(tdb + c0);
    float4 w0 = specscale(sp4[0], td4[0], ev);
    float4 w1 = specscale(sp4[1], td4[1], ev);
    float4 w2 = specscale(sp4[2], td4[2], ev);
    float4 w3 = specscale(sp4[3], td4[3], ev);
    *(uint4*)(sS + r * 136 + c0)     = cvt16(w0, w1, 1.f);
    *(uint4*)(sS + r * 136 + c0 + 8) = cvt16(w2, w3, 1.f);
  }
  __syncthreads();
  int w = rfl(tid >> 5), lane = tid & 31;
  int row0 = rfl(rb * 128 + w * 16);
  int i = lane >> 1, ch = (lane & 1) * 16;
  int r = row0 + i;
  float msk = (r < NPG) ? 1.f : 0.f;
  const float* ebase = evecs + ((size_t)b * NPG + ((r < NPG) ? r : 0)) * KK + ch;
  f32x8 z = {0, 0, 0, 0, 0, 0, 0, 0};
  f32x8 acc[8] = {z, z, z, z, z, z, z, z};
#pragma unroll
  for (int kk = 0; kk < 4; ++kk) {
    stage16(sA[w] + i * 40 + ch, ebase + kk * 32, msk);
    bf16x16 a = frag_a_lds(sA[w], 40);
#pragma unroll
    for (int t = 0; t < 8; ++t)
      acc[t] = WMMA_BF16(a, frag_b_lds(sS + kk * 32 * 136 + t * 16, 136), acc[t]);
  }
  int nc = lane & 15, hi = (lane < 16) ? 0 : 8;
  float* orow = hd + ((size_t)b * NPG + row0 + hi) * CC + nc;
  if (row0 + 16 <= NPG) {                      // scalar branch: full tile fast path
#pragma unroll
    for (int t = 0; t < 8; ++t)
#pragma unroll
      for (int j = 0; j < 8; ++j) orow[(size_t)j * CC + t * 16] = acc[t][j];
  } else {
#pragma unroll
    for (int t = 0; t < 8; ++t)
#pragma unroll
      for (int j = 0; j < 8; ++j)
        if (row0 + j + hi < NPG) orow[(size_t)j * CC + t * 16] = acc[t][j];
  }
}

// ------------- sparse complex gradient scatter: one wave per edge ------------------------
__global__ void __launch_bounds__(256) k_scatter(const int* __restrict__ rows,
                                                 const int* __restrict__ cols,
                                                 const float* __restrict__ vre,
                                                 const float* __restrict__ vim,
                                                 const float* __restrict__ hd,
                                                 float* __restrict__ gre,
                                                 float* __restrict__ gim) {
  int e = blockIdx.x * 8 + (threadIdx.x >> 5);
  if (e >= EE) return;
  int lane = threadIdx.x & 31;
  int r = rows[e], c = cols[e];
  float ar = vre[e], ai = vim[e];
  float4 xg = ((const float4*)hd)[(size_t)c * 32 + lane];
  float* pr = gre + (size_t)r * CC + lane * 4;
  float* pi = gim + (size_t)r * CC + lane * 4;
  atomAddF(pr + 0, ar * xg.x); atomAddF(pr + 1, ar * xg.y);
  atomAddF(pr + 2, ar * xg.z); atomAddF(pr + 3, ar * xg.w);
  atomAddF(pi + 0, ai * xg.x); atomAddF(pi + 1, ai * xg.y);
  atomAddF(pi + 2, ai * xg.z); atomAddF(pi + 3, ai * xg.w);
}

// ------------- g_feat = tanh(g_re*(g_re@A^T) + g_im*(g_im@A^T)) -> in-place into gre -----
__global__ void __launch_bounds__(256) k_gfeat(const float* __restrict__ A,
                                               float* __restrict__ gre,
                                               const float* __restrict__ gim) {
  __shared__ alignas(16) unsigned short sAt[KK * 136];
  __shared__ alignas(16) unsigned short sT[8][16 * 40];
  int tid = threadIdx.x;
  // sAt[k][c] = A[c][k]  (B = A^T); strided gather, packed b128 stores
#pragma unroll
  for (int pass = 0; pass < 4; ++pass) {
    int k = pass * 32 + (tid >> 3);
    int c0 = (tid & 7) * 16;
    unsigned int wv[8];
#pragma unroll
    for (int q = 0; q < 8; ++q)
      wv[q] = pk2(A[(size_t)(c0 + 2 * q) * KK + k], A[(size_t)(c0 + 2 * q + 1) * KK + k]);
    uint4 u0 = {wv[0], wv[1], wv[2], wv[3]}, u1 = {wv[4], wv[5], wv[6], wv[7]};
    *(uint4*)(sAt + k * 136 + c0) = u0;
    *(uint4*)(sAt + k * 136 + c0 + 8) = u1;
  }
  __syncthreads();
  int w = rfl(tid >> 5), lane = tid & 31;
  int row0 = rfl(blockIdx.x * 128 + w * 16);
  int i = lane >> 1, ch = (lane & 1) * 16;
  int r = row0 + i;
  float msk = (r < NN) ? 1.f : 0.f;
  size_t rowoff = (size_t)((r < NN) ? r : 0) * CC + ch;
  f32x8 z = {0, 0, 0, 0, 0, 0, 0, 0};
  f32x8 aR[8] = {z, z, z, z, z, z, z, z};
  f32x8 aI[8] = {z, z, z, z, z, z, z, z};
#pragma unroll
  for (int kk = 0; kk < 4; ++kk) {
    stage16(sT[w] + i * 40 + ch, gre + rowoff + kk * 32, msk);
    bf16x16 a = frag_a_lds(sT[w], 40);
#pragma unroll
    for (int t = 0; t < 8; ++t)
      aR[t] = WMMA_BF16(a, frag_b_lds(sAt + kk * 32 * 136 + t * 16, 136), aR[t]);
  }
#pragma unroll
  for (int kk = 0; kk < 4; ++kk) {
    stage16(sT[w] + i * 40 + ch, gim + rowoff + kk * 32, msk);
    bf16x16 a = frag_a_lds(sT[w], 40);
#pragma unroll
    for (int t = 0; t < 8; ++t)
      aI[t] = WMMA_BF16(a, frag_b_lds(sAt + kk * 32 * 136 + t * 16, 136), aI[t]);
  }
  int nc = lane & 15, hi = (lane < 16) ? 0 : 8;
  float* pr = gre + ((size_t)row0 + hi) * CC + nc;
  const float* pi = gim + ((size_t)row0 + hi) * CC + nc;
  if (row0 + 16 <= NN) {
#pragma unroll
    for (int t = 0; t < 8; ++t)
#pragma unroll
      for (int j = 0; j < 8; ++j) {
        size_t o = (size_t)j * CC + t * 16;
        pr[o] = tanhf(pr[o] * aR[t][j] + pi[o] * aI[t][j]);
      }
  } else {
#pragma unroll
    for (int t = 0; t < 8; ++t)
#pragma unroll
      for (int j = 0; j < 8; ++j)
        if (row0 + j + hi < NN) {
          size_t o = (size_t)j * CC + t * 16;
          pr[o] = tanhf(pr[o] * aR[t][j] + pi[o] * aI[t][j]);
        }
  }
}

// ------------- MLP layer1: hid = relu([h|hd|gf] @ W1 + b1), K=384 chunked ----------------
__global__ void __launch_bounds__(256) k_mlp1(const float* __restrict__ h,
                                              const float* __restrict__ hd,
                                              const float* __restrict__ gf,
                                              const float* __restrict__ W1b,
                                              const float* __restrict__ b1b,
                                              float* __restrict__ hid) {
  __shared__ alignas(16) unsigned short sB[32 * 136];
  __shared__ alignas(16) unsigned short sA[8][16 * 40];
  int tid = threadIdx.x, w = rfl(tid >> 5), lane = tid & 31;
  int row0 = rfl(blockIdx.x * 128 + w * 16);
  int i = lane >> 1, ch = (lane & 1) * 16;
  int r = row0 + i;
  float msk = (r < NN) ? 1.f : 0.f;
  size_t rowoff = (size_t)((r < NN) ? r : 0) * CC + ch;
  f32x8 z = {0, 0, 0, 0, 0, 0, 0, 0};
  f32x8 acc[8] = {z, z, z, z, z, z, z, z};
  for (int kk = 0; kk < 12; ++kk) {
    {
      int rr = tid >> 3;                 // 0..31
      int c0 = (tid & 7) * 16;
      stage16(sB + rr * 136 + c0, &W1b[(size_t)(kk * 32 + rr) * CC + c0], 1.f);
    }
    __syncthreads();
    const float* src = (kk < 4) ? (h + kk * 32)
                                : ((kk < 8) ? (hd + (kk - 4) * 32) : (gf + (kk - 8) * 32));
    stage16(sA[w] + i * 40 + ch, src + rowoff, msk);
    bf16x16 a = frag_a_lds(sA[w], 40);
#pragma unroll
    for (int t = 0; t < 8; ++t)
      acc[t] = WMMA_BF16(a, frag_b_lds(sB + t * 16, 136), acc[t]);
    __syncthreads();
  }
  int nc = lane & 15, hi = (lane < 16) ? 0 : 8;
  float* orow = hid + ((size_t)row0 + hi) * CC + nc;
  if (row0 + 16 <= NN) {
#pragma unroll
    for (int t = 0; t < 8; ++t) {
      float bv = b1b[t * 16 + nc];
#pragma unroll
      for (int j = 0; j < 8; ++j)
        orow[(size_t)j * CC + t * 16] = fmaxf(acc[t][j] + bv, 0.f);
    }
  } else {
#pragma unroll
    for (int t = 0; t < 8; ++t) {
      float bv = b1b[t * 16 + nc];
#pragma unroll
      for (int j = 0; j < 8; ++j)
        if (row0 + j + hi < NN) orow[(size_t)j * CC + t * 16] = fmaxf(acc[t][j] + bv, 0.f);
    }
  }
}

// ------------- MLP layer2 + residual: h += hid @ W2 + b2 --------------------------------
__global__ void __launch_bounds__(256) k_mlp2(const float* __restrict__ hid,
                                              const float* __restrict__ W2b,
                                              const float* __restrict__ b2b,
                                              float* __restrict__ h) {
  __shared__ alignas(16) unsigned short sW[KK * 136];
  __shared__ alignas(16) unsigned short sA[8][16 * 40];
  int tid = threadIdx.x;
#pragma unroll
  for (int pass = 0; pass < 4; ++pass) {
    int r = pass * 32 + (tid >> 3);
    int c0 = (tid & 7) * 16;
    stage16(sW + r * 136 + c0, &W2b[(size_t)r * CC + c0], 1.f);
  }
  __syncthreads();
  int w = rfl(tid >> 5), lane = tid & 31;
  int row0 = rfl(blockIdx.x * 128 + w * 16);
  int i = lane >> 1, ch = (lane & 1) * 16;
  int r = row0 + i;
  float msk = (r < NN) ? 1.f : 0.f;
  size_t rowoff = (size_t)((r < NN) ? r : 0) * CC + ch;
  f32x8 z = {0, 0, 0, 0, 0, 0, 0, 0};
  f32x8 acc[8] = {z, z, z, z, z, z, z, z};
#pragma unroll
  for (int kk = 0; kk < 4; ++kk) {
    stage16(sA[w] + i * 40 + ch, hid + rowoff + kk * 32, msk);
    bf16x16 a = frag_a_lds(sA[w], 40);
#pragma unroll
    for (int t = 0; t < 8; ++t)
      acc[t] = WMMA_BF16(a, frag_b_lds(sW + kk * 32 * 136 + t * 16, 136), acc[t]);
  }
  int nc = lane & 15, hi = (lane < 16) ? 0 : 8;
  float* orow = h + ((size_t)row0 + hi) * CC + nc;
  if (row0 + 16 <= NN) {
#pragma unroll
    for (int t = 0; t < 8; ++t) {
      float bv = b2b[t * 16 + nc];
#pragma unroll
      for (int j = 0; j < 8; ++j) {
        size_t o = (size_t)j * CC + t * 16;
        orow[o] = orow[o] + acc[t][j] + bv;
      }
    }
  } else {
#pragma unroll
    for (int t = 0; t < 8; ++t) {
      float bv = b2b[t * 16 + nc];
#pragma unroll
      for (int j = 0; j < 8; ++j)
        if (row0 + j + hi < NN) {
          size_t o = (size_t)j * CC + t * 16;
          orow[o] = orow[o] + acc[t][j] + bv;
        }
    }
  }
}

// ------------- head layer1: z = relu([h | gs[batch]] @ Wh1 + bh1), K=139 (zero-padded) ---
__global__ void __launch_bounds__(256) k_head1(const float* __restrict__ h,
                                               const float* __restrict__ gs,
                                               const int* __restrict__ batch,
                                               const float* __restrict__ Wh1,
                                               const float* __restrict__ bh1,
                                               float* __restrict__ zo) {
  __shared__ alignas(16) unsigned short sW[160 * 72];
  __shared__ alignas(16) unsigned short sA[8][16 * 40];
  int tid = threadIdx.x;
#pragma unroll
  for (int pass = 0; pass < 3; ++pass) {
    int r = pass * 64 + (tid >> 2);      // 0..191
    if (r < 160) {
      int c0 = (tid & 3) * 16;
      if (r < CC + GSD) {
        stage16(sW + r * 72 + c0, &Wh1[(size_t)r * 64 + c0], 1.f);
      } else {
        uint4 zz = {0, 0, 0, 0};
        *(uint4*)(sW + r * 72 + c0) = zz;
        *(uint4*)(sW + r * 72 + c0 + 8) = zz;
      }
    }
  }
  __syncthreads();
  int w = rfl(tid >> 5), lane = tid & 31;
  int row0 = rfl(blockIdx.x * 128 + w * 16);
  int i = lane >> 1, ch = (lane & 1) * 16;
  int r = row0 + i;
  bool v = r < NN;
  float msk = v ? 1.f : 0.f;
  size_t rowoff = (size_t)(v ? r : 0) * CC + ch;
  f32x8 z = {0, 0, 0, 0, 0, 0, 0, 0};
  f32x8 acc[4] = {z, z, z, z};
#pragma unroll
  for (int kk = 0; kk < 5; ++kk) {
    if (kk < 4) {
      stage16(sA[w] + i * 40 + ch, h + rowoff + kk * 32, msk);
    } else {
      int g = v ? batch[r] : 0;
#pragma unroll
      for (int q = 0; q < 16; ++q) {
        int jj = ch + q;
        float mk2 = (v && jj < GSD) ? 1.f : 0.f;
        int src = g * GSD + ((jj < GSD) ? jj : (GSD - 1));
        sA[w][i * 40 + jj] = bfbits(gs[src] * mk2);
      }
    }
    bf16x16 a = frag_a_lds(sA[w], 40);
#pragma unroll
    for (int t = 0; t < 4; ++t)
      acc[t] = WMMA_BF16(a, frag_b_lds(sW + kk * 32 * 72 + t * 16, 72), acc[t]);
  }
  int nc = lane & 15, hi = (lane < 16) ? 0 : 8;
  float* orow = zo + ((size_t)row0 + hi) * 64 + nc;
  if (row0 + 16 <= NN) {
#pragma unroll
    for (int t = 0; t < 4; ++t) {
      float bv = bh1[t * 16 + nc];
#pragma unroll
      for (int j = 0; j < 8; ++j)
        orow[(size_t)j * 64 + t * 16] = fmaxf(acc[t][j] + bv, 0.f);
    }
  } else {
#pragma unroll
    for (int t = 0; t < 4; ++t) {
      float bv = bh1[t * 16 + nc];
#pragma unroll
      for (int j = 0; j < 8; ++j)
        if (row0 + j + hi < NN) orow[(size_t)j * 64 + t * 16] = fmaxf(acc[t][j] + bv, 0.f);
    }
  }
}

// ------------- head layer2: out = z @ Wh2 + bh2, [64 x 5] (scalar, mem-bound) ------------
__global__ void __launch_bounds__(256) k_head2(const float* __restrict__ zi,
                                               const float* __restrict__ Wh2,
                                               const float* __restrict__ bh2,
                                               float* __restrict__ out) {
  int idx = blockIdx.x * blockDim.x + threadIdx.x;
  if (idx >= NN * NCLF) return;
  int n = idx / NCLF, q = idx - n * NCLF;
  float s = bh2[q];
  const float* zr = zi + (size_t)n * 64;
#pragma unroll
  for (int j = 0; j < 64; ++j) s += zr[j] * Wh2[j * NCLF + q];
  out[idx] = s;
}

extern "C" void kernel_launch(void* const* d_in, const int* in_sizes, int n_in,
                              void* d_out, int out_size, void* d_ws, size_t ws_size,
                              hipStream_t stream) {
  (void)in_sizes; (void)n_in; (void)out_size; (void)ws_size;
  const float* x       = (const float*)d_in[0];
  const float* evals   = (const float*)d_in[1];
  const float* evecs   = (const float*)d_in[2];
  const float* mass    = (const float*)d_in[3];
  const float* gs      = (const float*)d_in[4];
  const int*   batch   = (const int*)d_in[5];
  const int*   grows   = (const int*)d_in[6];
  const int*   gcols   = (const int*)d_in[7];
  const float* vre     = (const float*)d_in[8];
  const float* vim     = (const float*)d_in[9];
  // d_in[10] = grad_n (only feeds an unused nnz statistic)
  const float* W_embed = (const float*)d_in[11];
  const float* b_embed = (const float*)d_in[12];
  const float* t_diff  = (const float*)d_in[13];
  const float* A_grad  = (const float*)d_in[14];
  const float* W1      = (const float*)d_in[15];
  const float* b1      = (const float*)d_in[16];
  const float* W2      = (const float*)d_in[17];
  const float* b2      = (const float*)d_in[18];
  const float* Wh1     = (const float*)d_in[19];
  const float* bh1     = (const float*)d_in[20];
  const float* Wh2     = (const float*)d_in[21];
  const float* bh2     = (const float*)d_in[22];
  float* out = (float*)d_out;

  // workspace: h | h_diff | g_re | g_im | spec  (~205 MB)
  float* ws   = (float*)d_ws;
  float* h    = ws;
  float* hd   = h   + (size_t)NN * CC;
  float* gre  = hd  + (size_t)NN * CC;
  float* gim  = gre + (size_t)NN * CC;
  float* spec = gim + (size_t)NN * CC;

  dim3 b256(256);
  k_embed<<<(NN * CC + 255) / 256, b256, 0, stream>>>(x, W_embed, b_embed, h);

  for (int blk = 0; blk < NBLKS; ++blk) {
    int fillN = 2 * NN * CC + NB * KK * CC;   // g_re, g_im, spec (contiguous)
    k_fill0<<<(fillN + 255) / 256, b256, 0, stream>>>(gre, fillN);
    k_to_basis<<<512, dim3(128), 0, stream>>>(h, evecs, mass, spec);
    k_from_basis<<<NB * RBPG, b256, 0, stream>>>(spec, evecs, evals, t_diff + blk * CC, hd);
    k_scatter<<<EE / 8, b256, 0, stream>>>(grows, gcols, vre, vim, hd, gre, gim);
    k_gfeat<<<RBALL, b256, 0, stream>>>(A_grad + (size_t)blk * CC * CC, gre, gim);
    k_mlp1<<<RBALL, b256, 0, stream>>>(h, hd, gre, W1 + (size_t)blk * 3 * CC * CC,
                                       b1 + blk * CC, gim);
    k_mlp2<<<RBALL, b256, 0, stream>>>(gim, W2 + (size_t)blk * CC * CC, b2 + blk * CC, h);
  }

  k_head1<<<RBALL, b256, 0, stream>>>(h, gs, batch, Wh1, bh1, hd);
  k_head2<<<(NN * NCLF + 255) / 256, b256, 0, stream>>>(hd, Wh2, bh2, out);
}
